// PermutationModule_18537078849804
// MI455X (gfx1250) — matrix operand order repaired and checked by reading, not compiled, not run on hardware
//
#include <hip/hip_runtime.h>

typedef __attribute__((ext_vector_type(16))) __bf16 v16bf;
typedef __attribute__((ext_vector_type(8)))  float  v8f;

#define BB   8
#define NN   256
#define DD   1024
#define PP   512
#define ROWS (BB * NN)                   // 2048 rows of x
#define ROW_TILES (ROWS / 16)            // 128
#define COL_STRIPS_PER_MAT (PP / 64)     // 8 strips of 64 columns
#define TOT_STRIPS (ROW_TILES * 4 * COL_STRIPS_PER_MAT)  // 4096 waves
#define WAVES_PER_BLOCK 8

// ---------------------------------------------------------------------------
// Init start/end score outputs with their output biases (atomicAdd target).
// ---------------------------------------------------------------------------
__global__ void init_scores(float* __restrict__ out,
                            const float* __restrict__ bs2,
                            const float* __restrict__ be2) {
    int t = blockIdx.x * blockDim.x + threadIdx.x;
    if (t < ROWS) {
        out[(size_t)BB * NN * NN + t]        = bs2[0];
        out[(size_t)BB * NN * NN + ROWS + t] = be2[0];
    }
}

// ---------------------------------------------------------------------------
// Fused projection GEMM, register-blocked: one wave computes a 16x64 strip of
// C = x * W^T for one of {Wl, Wr, Ws1, We1}.  One A fragment feeds 4 WMMAs
// (4x A reuse, conversion amortized).  Wl/Wr strips write left/right to the
// workspace; Ws1/We1 strips apply relu, scale by Ws2/We2, reduce all 64
// columns and atomically accumulate into the start/end score outputs.
// ---------------------------------------------------------------------------
__global__ __launch_bounds__(256) void proj_gemm(
    const float* __restrict__ x,
    const float* __restrict__ Wl, const float* __restrict__ bl,
    const float* __restrict__ Wr,
    const float* __restrict__ Ws1, const float* __restrict__ bs1,
    const float* __restrict__ Ws2,
    const float* __restrict__ We1, const float* __restrict__ be1,
    const float* __restrict__ We2,
    float* __restrict__ left, float* __restrict__ right,
    float* __restrict__ out)
{
    const int wave  = threadIdx.x >> 5;
    const int lane  = threadIdx.x & 31;
    const int strip = blockIdx.x * WAVES_PER_BLOCK + wave;

    const int rowTile  = strip & (ROW_TILES - 1);
    const int colStrip = strip / ROW_TILES;
    const int mat      = colStrip >> 3;             // 0=Wl 1=Wr 2=Ws1 3=We1
    const int colBase  = (colStrip & 7) * 64;
    const int rowBase  = rowTile * 16;

    const float* W = (mat == 0) ? Wl : (mat == 1) ? Wr : (mat == 2) ? Ws1 : We1;

    const int half = lane >> 4;       // which 16-lane half of the wave
    const int l15  = lane & 15;
    const int arow = rowBase + l15;   // A-matrix row this lane supplies

    v8f c0 = {}, c1 = {}, c2 = {}, c3 = {};
    for (int k = 0; k < DD; k += 32) {
        // A 16x32 bf16 layout (matches B): lanes 0-15 K=0..15,
        // lanes 16-31 K=16..31; VGPR v packs a K pair.
        v16bf a;
        const float* xrow = x + (size_t)arow * DD + k + 16 * half;
        #pragma unroll
        for (int v = 0; v < 8; ++v) {
            int k0 = 2 * v;
            float2 f = *(const float2*)(xrow + k0);
            a[2 * v]     = (__bf16)f.x;
            a[2 * v + 1] = (__bf16)f.y;
        }
        // B 32x16 bf16: lane holds one column (weight row p); lanes 0-15
        // K=0..15, lanes 16-31 K=16..31 -> contiguous, coalesced float4s.
        #pragma unroll
        for (int t = 0; t < 4; ++t) {
            v16bf bm;
            const int bcol = colBase + 16 * t + l15;
            const float* wrow = W + (size_t)bcol * DD + k + 16 * half;
            #pragma unroll
            for (int v = 0; v < 4; ++v) {
                float4 f = *(const float4*)(wrow + 4 * v);
                bm[4 * v]     = (__bf16)f.x;
                bm[4 * v + 1] = (__bf16)f.y;
                bm[4 * v + 2] = (__bf16)f.z;
                bm[4 * v + 3] = (__bf16)f.w;
            }
            if (t == 0) c0 = __builtin_amdgcn_wmma_f32_16x16x32_bf16(
                                 false, a, false, bm, (short)0, c0, false, false);
            if (t == 1) c1 = __builtin_amdgcn_wmma_f32_16x16x32_bf16(
                                 false, a, false, bm, (short)0, c1, false, false);
            if (t == 2) c2 = __builtin_amdgcn_wmma_f32_16x16x32_bf16(
                                 false, a, false, bm, (short)0, c2, false, false);
            if (t == 3) c3 = __builtin_amdgcn_wmma_f32_16x16x32_bf16(
                                 false, a, false, bm, (short)0, c3, false, false);
        }
    }

    // C/D layout: VGPR r holds row rowBase+r (lanes 0-15) / rowBase+r+8
    // (lanes 16-31); column = colBase + 16*t + (lane & 15).
    v8f* acc[4] = { &c0, &c1, &c2, &c3 };
    if (mat <= 1) {
        float* dst = (mat == 0) ? left : right;
        float bias[4];
        #pragma unroll
        for (int t = 0; t < 4; ++t)
            bias[t] = (mat == 0) ? bl[colBase + 16 * t + l15] : 0.0f;
        #pragma unroll
        for (int r = 0; r < 8; ++r) {
            const int row = rowBase + r + 8 * half;
            #pragma unroll
            for (int t = 0; t < 4; ++t)
                dst[(size_t)row * PP + colBase + 16 * t + l15] = (*acc[t])[r] + bias[t];
        }
    } else {
        const float* b1 = (mat == 2) ? bs1 : be1;
        const float* w2 = (mat == 2) ? Ws2 : We2;
        float* score = out + (size_t)BB * NN * NN + ((mat == 3) ? ROWS : 0);
        float bias[4], wv[4];
        #pragma unroll
        for (int t = 0; t < 4; ++t) {
            bias[t] = b1[colBase + 16 * t + l15];
            wv[t]   = w2[colBase + 16 * t + l15];
        }
        #pragma unroll
        for (int r = 0; r < 8; ++r) {
            // combine the 4 column tiles of this row, then reduce the 16
            // lanes of the half-wave -> one atomic per row per wave
            float h = 0.0f;
            #pragma unroll
            for (int t = 0; t < 4; ++t)
                h += fmaxf((*acc[t])[r] + bias[t], 0.0f) * wv[t];
            h += __shfl_xor(h, 1, 32);
            h += __shfl_xor(h, 2, 32);
            h += __shfl_xor(h, 4, 32);
            h += __shfl_xor(h, 8, 32);
            if (l15 == 0)
                atomicAdd(&score[rowBase + r + 8 * half], h);
        }
    }
}

// ---------------------------------------------------------------------------
// Bigram: out[b,i,j] = sum_p relu(left[b,j,p] + right[b,i,p]) * Wo[p] + bo.
// One block per (b,i); right row + Wo staged in LDS; one wave per j with a
// wave32 shuffle-tree reduction over P.
// ---------------------------------------------------------------------------
__global__ __launch_bounds__(256) void bigram_kernel(
    const float* __restrict__ left, const float* __restrict__ right,
    const float* __restrict__ Wo, const float* __restrict__ bo,
    float* __restrict__ out)
{
    __shared__ float rbuf[PP];
    __shared__ float wobuf[PP];
    const int bi = blockIdx.x;              // b*N + i
    const int t  = threadIdx.x;
    rbuf[t]        = right[(size_t)bi * PP + t];
    rbuf[t + 256]  = right[(size_t)bi * PP + t + 256];
    wobuf[t]       = Wo[t];
    wobuf[t + 256] = Wo[t + 256];
    __syncthreads();

    const int wave = t >> 5, lane = t & 31;
    const int b = bi / NN;
    const float bo0 = bo[0];

    for (int j = wave; j < NN; j += 8) {
        const float* lrow = left + ((size_t)b * NN + j) * PP;
        __builtin_prefetch(lrow + lane, 0, 1);   // global_prefetch_b8
        float acc = 0.0f;
        #pragma unroll 4
        for (int p = lane; p < PP; p += 32)
            acc += fmaxf(lrow[p] + rbuf[p], 0.0f) * wobuf[p];
        acc += __shfl_xor(acc, 16, 32);
        acc += __shfl_xor(acc, 8, 32);
        acc += __shfl_xor(acc, 4, 32);
        acc += __shfl_xor(acc, 2, 32);
        acc += __shfl_xor(acc, 1, 32);
        if (lane == 0)
            out[(size_t)bi * NN + j] = acc + bo0;
    }
}

// ---------------------------------------------------------------------------
extern "C" void kernel_launch(void* const* d_in, const int* in_sizes, int n_in,
                              void* d_out, int out_size, void* d_ws, size_t ws_size,
                              hipStream_t stream) {
    const float* x   = (const float*)d_in[0];
    const float* Wl  = (const float*)d_in[1];
    const float* bl  = (const float*)d_in[2];
    const float* Wr  = (const float*)d_in[3];
    const float* Wo  = (const float*)d_in[4];
    const float* bo  = (const float*)d_in[5];
    const float* Ws1 = (const float*)d_in[6];
    const float* bs1 = (const float*)d_in[7];
    const float* Ws2 = (const float*)d_in[8];
    const float* bs2 = (const float*)d_in[9];
    const float* We1 = (const float*)d_in[10];
    const float* be1 = (const float*)d_in[11];
    const float* We2 = (const float*)d_in[12];
    const float* be2 = (const float*)d_in[13];

    float* out    = (float*)d_out;
    float* left   = (float*)d_ws;                 // [ROWS, PP]  4 MB
    float* rightb = left + (size_t)ROWS * PP;     // [ROWS, PP]  4 MB

    init_scores<<<(ROWS + 255) / 256, 256, 0, stream>>>(out, bs2, be2);
    proj_gemm<<<TOT_STRIPS / WAVES_PER_BLOCK, 256, 0, stream>>>(
        x, Wl, bl, Wr, Ws1, bs1, Ws2, We1, be1, We2, left, rightb, out);
    bigram_kernel<<<ROWS, 256, 0, stream>>>(left, rightb, Wo, bo, out);
}
